// BetaVAEMark7Encoder_34376918237628
// MI455X (gfx1250) — matrix operand order repaired and checked by reading, not compiled, not run on hardware
//
#include <hip/hip_runtime.h>

// ---------------------------------------------------------------------------
// Fused BetaVAE encoder for MI455X (gfx1250, wave32, WMMA, 320KB LDS/WGP).
// One workgroup (256 threads = 8 waves) processes one image entirely in LDS.
// Activations stored HWC (channel-last, channels padded to 8/16/32) with zero
// halos so WMMA A-fragment gathers are branchless contiguous ds_load_b128.
// All convs are implicit-GEMM via v_wmma_f32_16x16x32_bf16 (f32 accumulate).
// Per-N-tile bias and B-fragments are hoisted out of the M-tile loop.
// ---------------------------------------------------------------------------

typedef __bf16 bf16_t;
typedef __attribute__((ext_vector_type(16))) __bf16 v16bf;
typedef __attribute__((ext_vector_type(8)))  __bf16 v8bf;
typedef __attribute__((ext_vector_type(8)))  float  v8f;

#define WAVES 8
#define NTHREADS 256

// ---------------------------------------------------------------------------
// LDS layout (bf16, HWC). Lifetime-based reuse:
//   r1: X[102][34][8]=27744   then { P1[52][18][8]=7488 , H2[50][16][16]=12800 }
//   r2: H1[100][32][8]=25600  then { P2[12][10][16]=1920, H3[10][8][32]=2560, P3[2][4][32]=256 }
//   W : prepacked WMMA B-fragments, 21504 elems
// Total = (27744 + 25600 + 21504) * 2 B = 149,696 B -> 2 workgroups / WGP.
// ---------------------------------------------------------------------------
struct __align__(16) Smem {
    union {
        bf16_t X[102 * 34 * 8];
        struct { bf16_t P1[52 * 18 * 8]; bf16_t H2[50 * 16 * 16]; } s;
    } r1;
    union {
        bf16_t H1[100 * 32 * 8];
        struct { bf16_t P2[12 * 10 * 16]; bf16_t H3[10 * 8 * 32]; bf16_t P3[2 * 4 * 32]; } s;
    } r2;
    bf16_t W[21504];
};

// ---------------------------------------------------------------------------
// Prepack global fp32 weights [OC][IC][KH][KW] into LDS WMMA B-fragments.
// K-order: k = tap*ICP + ic (tap = ky*KW+kx, channels padded to ICP).
// Dense 32x16 B layout (matches A): lanes 0-15 elem e<8 -> K = ks*32+e, e>=8 -> K = ks*32+8+e, N=lane;
// lanes 16-31: same +8, N=lane-16.
// Storage: dst[((ks*NT + nt)*32 + lane)*16 + e]  (32B/lane, 16B-aligned).
// ---------------------------------------------------------------------------
template<int OC, int IC, int ICP, int TAPS, int KS, int NT>
__device__ void prepack(const float* __restrict__ wg, bf16_t* __restrict__ dst, int tid)
{
    constexpr int TOT = KS * NT * 512;
    for (int idx = tid; idx < TOT; idx += NTHREADS) {
        const int e    = idx & 15;
        const int lane = (idx >> 4) & 31;
        const int rest = idx >> 9;
        const int nt   = rest % NT;
        const int ks   = rest / NT;
        const int n    = nt * 16 + (lane & 15);
        const int kk   = ks * 32 + ((lane & 16) ? 8 : 0) + (e < 8 ? e : e + 8);  // padded K index
        const int tap  = kk / ICP;
        const int ic   = kk % ICP;
        float v = 0.0f;
        if (n < OC && tap < TAPS && ic < IC)
            v = wg[(n * IC + ic) * TAPS + tap];
        dst[idx] = (bf16_t)v;
    }
}

// ---------------------------------------------------------------------------
// Implicit-GEMM conv on LDS HWC buffers.
//   M = OH*OW (16 rows/tile), N = OC (16/tile), K = KH*KW*ICP (32/WMMA step).
// A-fragment (ISA 16-bit A layout): lane<16 elems0-7 -> K=kb..kb+7,
// elems8-15 -> K=kb+16..kb+23 (kb=ks*32); lane>=16: +8 on both runs.
// With HWC + ICP in {8,16,32}, each 8-run is one contiguous 16B LDS load.
// Input buffer has width IWP (= IW + 2*halo); 'SAME' convs read via the halo
// (always in bounds). K-padding taps clamped to last valid tap (B==0 there).
// nt loop unrolled: bias + all KS B-fragments hoisted into registers; inner
// mt loop = addr math + 2*KS ds_load_b128 + KS WMMA + store epilogue.
// Loops are wave-uniform -> EXEC all-ones at every WMMA.
// ---------------------------------------------------------------------------
template<int ICP, int KH, int KW, int SH, int SW, int IWP,
         int OH, int OW, int OC, int OHALO, bool LRELU>
__device__ void conv_wmma(const bf16_t* __restrict__ actin,
                          bf16_t* __restrict__ actout,
                          const bf16_t* __restrict__ wpack,
                          const float* __restrict__ bg, int tid)
{
    constexpr int M    = OH * OW;
    constexpr int MT   = (M + 15) / 16;
    constexpr int KP   = KH * KW * ICP;
    constexpr int KS   = (KP + 31) / 32;
    constexpr int NT   = (OC + 15) / 16;
    constexpr int OWP  = OW + 2 * OHALO;
    constexpr int TMAX = KH * KW - 1;

    const int lane = tid & 31;
    const int wave = tid >> 5;

#pragma unroll
    for (int nt = 0; nt < NT; ++nt) {
        // ---- hoisted per-N-tile state: bias + B fragments in registers ----
        const int   nn   = nt * 16 + (lane & 15);
        const float bias = (nn < OC) ? bg[nn] : 0.0f;
        v16bf bfrag[KS];
#pragma unroll
        for (int ks = 0; ks < KS; ++ks)
            bfrag[ks] = *(const v16bf*)(wpack + (((ks * NT + nt) * 32 + lane) << 4));

        for (int mt = wave; mt < MT; mt += WAVES) {
            const int m0 = mt * 16;
            int row = m0 + (lane & 15);
            if (row > M - 1) row = M - 1;           // only matters when M%16!=0
            const int oy    = row / OW;             // OW is a power of two
            const int ox    = row % OW;
            const int ibase = (oy * SH) * IWP + (ox * SW);

            v8f acc = {bias, bias, bias, bias, bias, bias, bias, bias};

#pragma unroll
            for (int ks = 0; ks < KS; ++ks) {
                const int kr0 = ks * 32 + ((lane & 16) ? 8 : 0);
                int tap0 = kr0 / ICP;        if (tap0 > TMAX) tap0 = TMAX;
                int tap1 = (kr0 + 16) / ICP; if (tap1 > TMAX) tap1 = TMAX;
                const int c0 = kr0 % ICP;
                const int c1 = (kr0 + 16) % ICP;
                const int a0 = (ibase + (tap0 / KW) * IWP + (tap0 % KW)) * ICP + c0;
                const int a1 = (ibase + (tap1 / KW) * IWP + (tap1 % KW)) * ICP + c1;

                const v8bf r0 = *(const v8bf*)(actin + a0);     // ds_load_b128
                const v8bf r1 = *(const v8bf*)(actin + a1);     // ds_load_b128
                const v16bf a = __builtin_shufflevector(
                    r0, r1, 0, 1, 2, 3, 4, 5, 6, 7, 8, 9, 10, 11, 12, 13, 14, 15);

                acc = __builtin_amdgcn_wmma_f32_16x16x32_bf16(
                          false, a, false, bfrag[ks], (short)0, acc, false, false);
            }

            // epilogue: lrelu + store bf16 HWC (lanes0-15 rows m0+r,
            // lanes16-31 rows m0+8+r, N = lane%16). Bias already in acc.
            if (nn < OC) {
                const int mbase = m0 + ((lane & 16) ? 8 : 0);
#pragma unroll
                for (int r = 0; r < 8; ++r) {
                    const int m = mbase + r;
                    if (m < M) {
                        float y = acc[r];
                        if (LRELU) y = fmaxf(y, 0.01f * y);   // slope<1 => max form
                        int oaddr;
                        if (OHALO) {
                            const int my = m / OW, mx = m % OW;  // OW power of two
                            oaddr = ((my + 1) * OWP + (mx + 1)) * OC + nn;
                        } else {
                            oaddr = m * OC + nn;
                        }
                        actout[oaddr] = (bf16_t)y;
                    }
                }
            }
        }
    }
}

// ---------------------------------------------------------------------------
__global__ __launch_bounds__(NTHREADS)
void betavae_encoder_kernel(
    const float* __restrict__ x,
    const float* __restrict__ w1,  const float* __restrict__ b1,
    const float* __restrict__ w2,  const float* __restrict__ b2,
    const float* __restrict__ w3,  const float* __restrict__ b3,
    const float* __restrict__ wp1, const float* __restrict__ bp1,
    const float* __restrict__ wp2, const float* __restrict__ bp2,
    const float* __restrict__ wp3, const float* __restrict__ bp3,
    const float* __restrict__ wmu, const float* __restrict__ bmu,
    const float* __restrict__ wlv, const float* __restrict__ blv,
    float* __restrict__ out, int batch)
{
    __shared__ Smem sm;
    const int tid = threadIdx.x;
    const int img = blockIdx.x;

    const v8bf z8 = {(bf16_t)0.f, (bf16_t)0.f, (bf16_t)0.f, (bf16_t)0.f,
                     (bf16_t)0.f, (bf16_t)0.f, (bf16_t)0.f, (bf16_t)0.f};

    // ---- phase 0: zero X region (covers halo + padded channels 6,7) ----
    {
        v8bf* p = (v8bf*)sm.r1.X;
        for (int i = tid; i < (102 * 34 * 8) / 8; i += NTHREADS) p[i] = z8;
    }
    __syncthreads();

    // ---- phase 1: stage x (CHW fp32 -> HWC bf16 interior) + prepack weights ----
    {
        const float* xin = x + (size_t)img * 19200;
        for (int ic = 0; ic < 6; ++ic) {
            const float* src = xin + ic * 3200;
            for (int t = tid; t < 800; t += NTHREADS) {
                const float4 v = ((const float4*)src)[t];   // global_load_b128
                const int s  = t * 4;
                const int iy = s >> 5, ix = s & 31;
                bf16_t* dst = sm.r1.X + ((iy + 1) * 34 + (ix + 1)) * 8 + ic;
                dst[0]  = (bf16_t)v.x;
                dst[8]  = (bf16_t)v.y;
                dst[16] = (bf16_t)v.z;
                dst[24] = (bf16_t)v.w;
            }
        }
        prepack<8,   6,  8,  9,  3, 1>(w1,  sm.W + 0,     tid);  // 1536
        prepack<8,   8,  8,  4,  1, 1>(wp1, sm.W + 1536,  tid);  //  512
        prepack<16,  8,  8,  9,  3, 1>(w2,  sm.W + 2048,  tid);  // 1536
        prepack<16, 16, 16, 10,  5, 1>(wp2, sm.W + 3584,  tid);  // 2560
        prepack<32, 16, 16,  9,  5, 2>(w3,  sm.W + 6144,  tid);  // 5120
        prepack<32, 32, 32, 10, 10, 2>(wp3, sm.W + 11264, tid);  // 10240
    }
    __syncthreads();

    // block 1: 3x3 same (6p8 -> 8) + lrelu, then 2x2/2 pool-conv (8 -> 8)
    conv_wmma<8, 3, 3, 1, 1, 34, 100, 32, 8, 0, true>(sm.r1.X, sm.r2.H1, sm.W + 0, b1, tid);
    __syncthreads();
    {   // zero P1 halo ring (disjoint from pool1 interior writes)
        v8bf* p1 = (v8bf*)sm.r1.s.P1;
        for (int idx = tid; idx < 52 * 18; idx += NTHREADS) {
            const int yy = idx / 18, xx = idx % 18;
            if (yy == 0 || yy == 51 || xx == 0 || xx == 17) p1[idx] = z8;
        }
        conv_wmma<8, 2, 2, 2, 2, 32, 50, 16, 8, 1, false>(sm.r2.H1, sm.r1.s.P1, sm.W + 1536, bp1, tid);
    }
    __syncthreads();
    // block 2
    conv_wmma<8, 3, 3, 1, 1, 18, 50, 16, 16, 0, true>(sm.r1.s.P1, sm.r1.s.H2, sm.W + 2048, b2, tid);
    __syncthreads();
    {   // zero P2 halo ring
        v8bf* p2 = (v8bf*)sm.r2.s.P2;
        for (int idx = tid; idx < 12 * 10; idx += NTHREADS) {
            const int yy = idx / 10, xx = idx % 10;
            if (yy == 0 || yy == 11 || xx == 0 || xx == 9) {
                p2[idx * 2] = z8; p2[idx * 2 + 1] = z8;
            }
        }
        conv_wmma<16, 5, 2, 5, 2, 16, 50, 8, 16, 1, false>(sm.r1.s.H2, sm.r2.s.P2, sm.W + 3584, bp2, tid);
    }
    __syncthreads();
    // block 3
    conv_wmma<16, 3, 3, 1, 1, 10, 10, 8, 32, 0, true>(sm.r2.s.P2, sm.r2.s.H3, sm.W + 6144, b3, tid);
    __syncthreads();
    conv_wmma<32, 5, 2, 5, 2, 8, 2, 4, 32, 0, false>(sm.r2.s.H3, sm.r2.s.P3, sm.W + 11264, bp3, tid);
    __syncthreads();

    // ---- heads: f[256] (NCHW flatten: i = c*8 + y*4 + x) -> mu[7], logvar[7] ----
    if (tid < 14) {
        const bool  isMu = tid < 7;
        const int   j    = isMu ? tid : tid - 7;
        const float* wv  = isMu ? (wmu + j * 256) : (wlv + j * 256);
        float s = isMu ? bmu[j] : blv[j];
#pragma unroll 4
        for (int i = 0; i < 256; ++i) {
            const int c = i >> 3, yx = i & 7;               // P3 is HWC
            s += (float)sm.r2.s.P3[yx * 32 + c] * wv[i];
        }
        if (!isMu) s = fminf(fmaxf(s, -5.0f), 0.0f);
        out[(isMu ? 0 : batch * 7) + img * 7 + j] = s;
    }
}

// ---------------------------------------------------------------------------
extern "C" void kernel_launch(void* const* d_in, const int* in_sizes, int n_in,
                              void* d_out, int out_size, void* d_ws, size_t ws_size,
                              hipStream_t stream)
{
    (void)n_in; (void)d_ws; (void)ws_size; (void)out_size;
    const float* x   = (const float*)d_in[0];
    const float* w1  = (const float*)d_in[1];
    const float* b1  = (const float*)d_in[2];
    const float* w2  = (const float*)d_in[3];
    const float* b2  = (const float*)d_in[4];
    const float* w3  = (const float*)d_in[5];
    const float* b3  = (const float*)d_in[6];
    const float* wp1 = (const float*)d_in[7];
    const float* bp1 = (const float*)d_in[8];
    const float* wp2 = (const float*)d_in[9];
    const float* bp2 = (const float*)d_in[10];
    const float* wp3 = (const float*)d_in[11];
    const float* bp3 = (const float*)d_in[12];
    const float* wmu = (const float*)d_in[13];
    const float* bmu = (const float*)d_in[14];
    const float* wlv = (const float*)d_in[15];
    const float* blv = (const float*)d_in[16];

    const int batch = in_sizes[0] / (6 * 100 * 32);   // 4096

    betavae_encoder_kernel<<<batch, NTHREADS, 0, stream>>>(
        x, w1, b1, w2, b2, w3, b3, wp1, bp1, wp2, bp2, wp3, bp3,
        wmu, bmu, wlv, blv, (float*)d_out, batch);
}
